// DecoderTemporal_15204184227933
// MI455X (gfx1250) — matrix operand order repaired and checked by reading, not compiled, not run on hardware
//
#include <hip/hip_runtime.h>
#include <hip/hip_bf16.h>

typedef __attribute__((ext_vector_type(16))) _Float16 v16h;
typedef __attribute__((ext_vector_type(8)))  float    v8f;

#define NPEDS 131072
#define SEQ   30
#define HDIM  64
#define EDIM  16
#define TW    20
#define WAVES 8
#define PW    16
#define ROWH  104   // hstage row stride in halfs (208B, multiple of 16B)

union Frag16 { v16h h; uint4 q[2]; unsigned int u[8]; };
union Acc8   { v8f  v; float f[8]; };

__device__ __forceinline__ float frcp(float x){ return __builtin_amdgcn_rcpf(x); }
__device__ __forceinline__ float fexp2(float x){ return __builtin_amdgcn_exp2f(x); }
__device__ __forceinline__ float sigm(float x){ return frcp(1.f + fexp2(x * -1.44269504f)); }
__device__ __forceinline__ float tanh_f(float x){
  float e = fexp2(x * -2.885390082f);          // exp(-2x)
  return (1.f - e) * frcp(1.f + e);
}
__device__ __forceinline__ unsigned int pack2(float x, float y){
  union { _Float16 h[2]; unsigned int u; } z;
  z.h[0] = (_Float16)x; z.h[1] = (_Float16)y; return z.u;
}

__device__ __forceinline__ v8f wmma(const Frag16& a, const Frag16& b, v8f c){
  return __builtin_amdgcn_wmma_f32_16x16x32_f16(false, a.h, false, b.h, (short)0, c, false, false);
}

// B fragment: VGPR r, lane l holds K = {2(r&3), 2(r&3)+1} + 8*(l>=16) + 16*(r>=4), N = l%16 (packed half2).
__device__ __forceinline__ Frag16 loadB(const unsigned int* base, int idx, int lane){
  Frag16 b;
  const unsigned int* p = base + ((idx * 32 + lane) << 3);
  b.q[0] = *(const uint4*)p;
  b.q[1] = *(const uint4*)(p + 4);
  return b;
}

// A fragment from a staged row (row = this lane's M): lanes<16 get K base +0, lanes>=16 +8;
// VGPRs 0-3 = halfs [off0,off0+8), VGPRs 4-7 = halfs [off0+16,off0+24).
__device__ __forceinline__ Frag16 loadA(const _Float16* row, int kc, int hi){
  Frag16 a;
  const _Float16* p = row + 32 * kc + 8 * hi;
  a.q[0] = *(const uint4*)p;
  a.q[1] = *(const uint4*)(p + 16);
  return a;
}

__global__ __launch_bounds__(256, 1)
void decoder_lstm_kernel(const float* __restrict__ traj_rel,
                         const float* __restrict__ h0,
                         const float* __restrict__ c0,
                         const float* __restrict__ W_ih,
                         const float* __restrict__ W_hh,
                         const float* __restrict__ b_ih,
                         const float* __restrict__ b_hh,
                         const float* __restrict__ W_emb,
                         const float* __restrict__ b_emb,
                         const float* __restrict__ W_pos,
                         const float* __restrict__ b_pos,
                         float* __restrict__ out)
{
  // B fragments (packed half2), shared by all waves in the WG
  __shared__ __align__(16) unsigned int Bg[16 * 3 * 32 * 8];  // gates: [W_hh | W_ih | 0], 48KB
  __shared__ __align__(16) unsigned int Be[2 * 32 * 8];       // embedding W_emb (K=40 pad 64)
  __shared__ __align__(16) unsigned int Bp[2 * 32 * 8];       // hidden2pos W_pos (2 valid cols)
  __shared__ float bsum[256];
  __shared__ float bembS[EDIM];
  __shared__ float bposS[2];
  // per-wave staging: A-source rows [h(64) | x(16) | zero-pad(16) | slack]
  __shared__ __align__(16) _Float16 hstage[WAVES][16][ROWH];
  // per-wave trajectory ring: half2(x,y) per (time, ped)
  __shared__ __align__(16) unsigned int ring[WAVES][TW][PW];
  // c-state in LDS (C-fragment layout, lane-major -> conflict-free b32)
  __shared__ float cstage[WAVES][4][8][32];

  const int tid  = threadIdx.x;
  const int wave = tid >> 5;
  const int lane = tid & 31;
  const int lo   = lane & 15;
  const int hi   = lane >> 4;
  const int pedbase = (blockIdx.x * WAVES + wave) * PW;

  // ---- cooperative weight packing ----
  for (int q = tid; q < 16 * 3 * 32 * 8; q += 256) {
    int r = q & 7, l = (q >> 3) & 31, tc = q >> 8;
    int kc = tc % 3, t = tc / 3;
    int n  = 16 * t + (l & 15);
    int k0 = 32 * kc + 2 * (r & 3) + 8 * (l >> 4) + 16 * (r >> 2);
    float v0 = 0.f, v1 = 0.f;
    int k = k0;
    v0 = (k < 64) ? W_hh[n * 64 + k] : ((k < 80) ? W_ih[n * 16 + (k - 64)] : 0.f);
    k = k0 + 1;
    v1 = (k < 64) ? W_hh[n * 64 + k] : ((k < 80) ? W_ih[n * 16 + (k - 64)] : 0.f);
    Bg[q] = pack2(v0, v1);
  }
  for (int q = tid; q < 2 * 32 * 8; q += 256) {
    int r = q & 7, l = (q >> 3) & 31, kc = q >> 8;
    int n  = l & 15;
    int k0 = 32 * kc + 2 * (r & 3) + 8 * (l >> 4) + 16 * (r >> 2);
    float v0 = (k0     < 40) ? W_emb[n * 40 + k0    ] : 0.f;
    float v1 = (k0 + 1 < 40) ? W_emb[n * 40 + k0 + 1] : 0.f;
    Be[q] = pack2(v0, v1);
    float p0 = (n < 2) ? W_pos[n * 64 + k0    ] : 0.f;
    float p1 = (n < 2) ? W_pos[n * 64 + k0 + 1] : 0.f;
    Bp[q] = pack2(p0, p1);
  }
  bsum[tid] = b_ih[tid] + b_hh[tid];
  if (tid < EDIM) bembS[tid] = b_emb[tid];
  if (tid < 2)    bposS[tid] = b_pos[tid];
  __syncthreads();

  // ---- per-wave state init ----
  #pragma unroll
  for (int t = 0; t < 4; t++)
    #pragma unroll
    for (int r = 0; r < 8; r++) {
      int ped = pedbase + r + 8 * hi;
      cstage[wave][t][r][lane] = c0[(size_t)ped * HDIM + 16 * t + lo];
      hstage[wave][r + 8 * hi][16 * t + lo] = (_Float16)h0[(size_t)ped * HDIM + 16 * t + lo];
    }
  { // persistent zero pad at K = 80..95
    uint4 z = {0u, 0u, 0u, 0u};
    *(uint4*)(&hstage[wave][lo][80 + 8 * hi]) = z;
  }
  // prime trajectory ring from traj_rel (20, NPEDS, 2)
  for (int i = 0; i < 10; i++) {
    int t = 10 * hi + i;
    const float* src = traj_rel + ((size_t)t * NPEDS + pedbase + lo) * 2;
    ring[wave][t][lo] = pack2(src[0], src[1]);
  }

  const _Float16* hrow = &hstage[wave][lo][0];

  // ---- embedding via WMMA: x = leaky(flat(16x40) @ W_emb^T + b_emb) -> hstage cols 64..79
  auto do_embed = [&](int st) {
    Frag16 a0, a1;
    #pragma unroll
    for (int r = 0; r < 8; r++) {
      int tt = (r < 4) ? (r + 4 * hi) : (8 + (r - 4) + 4 * hi);
      int sl = st + tt; if (sl >= TW) sl -= TW;
      a0.u[r] = ring[wave][sl][lo];
    }
    #pragma unroll
    for (int r = 0; r < 8; r++) {
      int sl = st + 16 + (r & 3); if (sl >= TW) sl -= TW;
      a1.u[r] = (hi == 0 && r < 4) ? ring[wave][sl][lo] : 0u;
    }
    Acc8 cx; float be = bembS[lo];
    #pragma unroll
    for (int r = 0; r < 8; r++) cx.f[r] = be;
    cx.v = wmma(a0, loadB(Be, 0, lane), cx.v);
    cx.v = wmma(a1, loadB(Be, 1, lane), cx.v);
    #pragma unroll
    for (int r = 0; r < 8; r++) {
      float v = cx.f[r];
      v = (v >= 0.f) ? v : 0.01f * v;
      hstage[wave][r + 8 * hi][64 + lo] = (_Float16)v;
    }
  };

  do_embed(0);
  int start = 0;

  #pragma unroll 1
  for (int s = 0; s < SEQ; s++) {
    // A fragments: [h | x | 0], K = 96 -> 3 chunks of K=32
    Frag16 a0 = loadA(hrow, 0, hi), a1 = loadA(hrow, 1, hi), a2 = loadA(hrow, 2, hi);

    // one gate-tile group at a time: keeps live VGPRs bounded (no vgpr-msb spill zone)
    #pragma unroll 1
    for (int t = 0; t < 4; t++) {
      Acc8 ci, cf, cg, co;
      float b0 = bsum[16 * t + lo],        b1 = bsum[16 * (t + 4) + lo];
      float b2 = bsum[16 * (t + 8) + lo],  b3 = bsum[16 * (t + 12) + lo];
      #pragma unroll
      for (int r = 0; r < 8; r++) { ci.f[r] = b0; cf.f[r] = b1; cg.f[r] = b2; co.f[r] = b3; }
      #pragma unroll
      for (int kc = 0; kc < 3; kc++) {
        const Frag16& a = (kc == 0) ? a0 : ((kc == 1) ? a1 : a2);
        ci.v = wmma(a, loadB(Bg, (t     ) * 3 + kc, lane), ci.v);
        cf.v = wmma(a, loadB(Bg, (t +  4) * 3 + kc, lane), cf.v);
        cg.v = wmma(a, loadB(Bg, (t +  8) * 3 + kc, lane), cg.v);
        co.v = wmma(a, loadB(Bg, (t + 12) * 3 + kc, lane), co.v);
      }
      float cold[8];
      #pragma unroll
      for (int r = 0; r < 8; r++) cold[r] = cstage[wave][t][r][lane];
      #pragma unroll
      for (int r = 0; r < 8; r++) {
        float iv = sigm(ci.f[r]), fv = sigm(cf.f[r]);
        float gv = tanh_f(cg.f[r]), ov = sigm(co.f[r]);
        float cc = fv * cold[r] + iv * gv;
        cstage[wave][t][r][lane] = cc;
        hstage[wave][r + 8 * hi][16 * t + lo] = (_Float16)(ov * tanh_f(cc));
      }
    }

    // rel_pos = sigmoid(h_new @ W_pos^T + b_pos); leaky(sigmoid(x)) == sigmoid(x)
    {
      Frag16 ha0 = loadA(hrow, 0, hi), ha1 = loadA(hrow, 1, hi);
      Acc8 cr; float bp = (lo < 2) ? bposS[lo] : 0.f;
      #pragma unroll
      for (int r = 0; r < 8; r++) cr.f[r] = bp;
      cr.v = wmma(ha0, loadB(Bp, 0, lane), cr.v);
      cr.v = wmma(ha1, loadB(Bp, 1, lane), cr.v);
      int slot = start;                   // overwrite oldest window entry
      if (lo < 2) {
        _Float16* rp = (_Float16*)&ring[wave][slot][0];
        #pragma unroll
        for (int r = 0; r < 8; r++) {
          float v = sigm(cr.f[r]);
          int m = r + 8 * hi;
          out[(size_t)s * NPEDS * 2 + (size_t)(pedbase + m) * 2 + lo] = v;
          rp[2 * m + lo] = (_Float16)v;
        }
      }
      start = (start + 1 == TW) ? 0 : start + 1;
    }

    do_embed(start);
  }
}

extern "C" void kernel_launch(void* const* d_in, const int* in_sizes, int n_in,
                              void* d_out, int out_size, void* d_ws, size_t ws_size,
                              hipStream_t stream) {
  // setup_inputs order: traj_abs(0, unused), traj_rel(1), h0(2), c0(3),
  // W_ih(4), W_hh(5), b_ih(6), b_hh(7), W_emb(8), b_emb(9), W_pos(10), b_pos(11)
  const float* traj_rel = (const float*)d_in[1];
  const float* h0    = (const float*)d_in[2];
  const float* c0    = (const float*)d_in[3];
  const float* W_ih  = (const float*)d_in[4];
  const float* W_hh  = (const float*)d_in[5];
  const float* b_ih  = (const float*)d_in[6];
  const float* b_hh  = (const float*)d_in[7];
  const float* W_emb = (const float*)d_in[8];
  const float* b_emb = (const float*)d_in[9];
  const float* W_pos = (const float*)d_in[10];
  const float* b_pos = (const float*)d_in[11];

  dim3 grid(NPEDS / (WAVES * PW));   // 1024 workgroups
  dim3 block(WAVES * 32);            // 8 wave32s
  hipLaunchKernelGGL(decoder_lstm_kernel, grid, block, 0, stream,
                     traj_rel, h0, c0, W_ih, W_hh, b_ih, b_hh,
                     W_emb, b_emb, W_pos, b_pos, (float*)d_out);
}
